// Get_fMRI_feature_9740985827380
// MI455X (gfx1250) — matrix operand order repaired and checked by reading, not compiled, not run on hardware
//
#include <hip/hip_runtime.h>
#include <hip/hip_bf16.h>

// MI455X / gfx1250, wave32. corr_b = normalize(Xc^T Xc) per batch; BN provably cancels.
// Compute-bound (34.4 GFLOP vs ~11us HBM) -> use the high-rate matrix path:
// bf16x3 split GEMM on V_WMMA_F32_16X16X32_BF16 (16384 FLOP/instr vs 2048 for f32 WMMA).

typedef __attribute__((ext_vector_type(8)))  __bf16 v8bf;
typedef __attribute__((ext_vector_type(16))) __bf16 v16bf;
typedef __attribute__((ext_vector_type(8)))  float  v8f;

#define B_   64
#define T_   1024
#define N_   512
#define TILE 128
#define KC   32    // one bf16-WMMA K depth per LDS slab
#define CST  40    // column stride (bf16 elems): 80B = 20 dwords -> conflict-free col fan-out, 16B aligned

union bfbits { __bf16 h; unsigned short u; };

static __device__ __forceinline__ unsigned short f2bf_bits(float x) {
  bfbits c; c.h = (__bf16)x; return c.u;
}
static __device__ __forceinline__ float bf2f_bits(unsigned short b) {
  bfbits c; c.u = b; return (float)c.h;
}
// split x0,x1 (adjacent K) into bf16 hi/lo planes, packed as (k)|(k+1)<<16
static __device__ __forceinline__ void split_pack(float x0, float x1,
                                                  unsigned int& hi, unsigned int& lo) {
  const unsigned short h0 = f2bf_bits(x0);
  const unsigned short h1 = f2bf_bits(x1);
  const unsigned short l0 = f2bf_bits(x0 - bf2f_bits(h0));
  const unsigned short l1 = f2bf_bits(x1 - bf2f_bits(h1));
  hi = (unsigned)h0 | ((unsigned)h1 << 16);
  lo = (unsigned)l0 | ((unsigned)l1 << 16);
}
// fragment = two contiguous 16B runs per lane (ISA 16-bit A/B VGPR layouts)
static __device__ __forceinline__ v16bf load_frag(const __bf16* p, int second_off) {
  const v8bf a = *reinterpret_cast<const v8bf*>(p);
  const v8bf b = *reinterpret_cast<const v8bf*>(p + second_off);
  v16bf f;
#pragma unroll
  for (int e = 0; e < 8; ++e) { f[e] = a[e]; f[e + 8] = b[e]; }
  return f;
}

// ---------------- Kernel 1: per-(b,n) column means ----------------
__global__ __launch_bounds__(256) void col_mean_kernel(const float* __restrict__ x,
                                                       float* __restrict__ mean) {
  int idx = blockIdx.x * 256 + threadIdx.x;        // [0, 64*512)
  int b = idx >> 9, n = idx & (N_ - 1);
  const float* p = x + (size_t)b * T_ * N_ + n;
  float s = 0.f;
#pragma unroll 8
  for (int t = 0; t < T_; ++t) s += p[(size_t)t * N_];
  mean[idx] = s * (1.0f / T_);
}

// ---------------- Kernel 2: G = Xc^T Xc via bf16x3 WMMA, tile 128x128 ----------------
__global__ __launch_bounds__(256) void gram_kernel(const float* __restrict__ x,
                                                   const float* __restrict__ mean,
                                                   float* __restrict__ g) {
  __shared__ __bf16 lAhi[TILE * CST];   // column-major: [col][k], 10KB each
  __shared__ __bf16 lAlo[TILE * CST];
  __shared__ __bf16 lBhi[TILE * CST];
  __shared__ __bf16 lBlo[TILE * CST];

  const int tid = threadIdx.x;
  const int b   = blockIdx.y;

  // blockIdx.x (0..9) -> upper-triangular tile (I<=J) of the 4x4 tile grid
  int rem = blockIdx.x, I = 0, rowlen = 4;
  while (rem >= rowlen) { rem -= rowlen; --rowlen; ++I; }
  const int J = I + rem;
  const int Ibase = I * TILE, Jbase = J * TILE;
  const bool mirror = (Ibase != Jbase);

  const float* srcA = x + (size_t)b * T_ * N_ + Ibase;
  const float* srcB = x + (size_t)b * T_ * N_ + Jbase;

  // loader: thread owns cols c4..c4+3, K pairs {kbase,kbase+1} and {kbase+16,kbase+17}
  const int c4    = (tid & 31) << 2;
  const int kbase = (tid >> 5) << 1;
  const float ma0 = mean[b * N_ + Ibase + c4 + 0], ma1 = mean[b * N_ + Ibase + c4 + 1];
  const float ma2 = mean[b * N_ + Ibase + c4 + 2], ma3 = mean[b * N_ + Ibase + c4 + 3];
  const float mb0 = mean[b * N_ + Jbase + c4 + 0], mb1 = mean[b * N_ + Jbase + c4 + 1];
  const float mb2 = mean[b * N_ + Jbase + c4 + 2], mb3 = mean[b * N_ + Jbase + c4 + 3];

  const int lane  = tid & 31;
  const int l15   = lane & 15;
  const int K0A   = (lane >> 4) << 3;              // A frag: lanes16-31 hold K+8
  const int K0B   = (lane >> 4) << 3;              // B frag: lanes16-31 hold K+8
  const int w     = tid >> 5;
  const int waveM = (w & 1) * 64;                  // 8 waves = 2 (M) x 4 (N)
  const int waveN = (w >> 1) * 32;

  v8f acc[4][2];
  const v8f vzero = {0.f, 0.f, 0.f, 0.f, 0.f, 0.f, 0.f, 0.f};
#pragma unroll
  for (int i = 0; i < 4; ++i)
#pragma unroll
    for (int j = 0; j < 2; ++j) acc[i][j] = vzero;

  for (int t0 = 0; t0 < T_; t0 += KC) {
    // ---- load, center, split to bf16 hi/lo, store transposed (column-major) ----
#pragma unroll
    for (int kq = 0; kq < 2; ++kq) {
      const int k0 = kbase + kq * 16;
      const float4 a0 = *reinterpret_cast<const float4*>(srcA + (size_t)(t0 + k0) * N_ + c4);
      const float4 a1 = *reinterpret_cast<const float4*>(srcA + (size_t)(t0 + k0 + 1) * N_ + c4);
      const float4 b0 = *reinterpret_cast<const float4*>(srcB + (size_t)(t0 + k0) * N_ + c4);
      const float4 b1 = *reinterpret_cast<const float4*>(srcB + (size_t)(t0 + k0 + 1) * N_ + c4);
      const float xa0[4] = {a0.x - ma0, a0.y - ma1, a0.z - ma2, a0.w - ma3};
      const float xa1[4] = {a1.x - ma0, a1.y - ma1, a1.z - ma2, a1.w - ma3};
      const float xb0[4] = {b0.x - mb0, b0.y - mb1, b0.z - mb2, b0.w - mb3};
      const float xb1[4] = {b1.x - mb0, b1.y - mb1, b1.z - mb2, b1.w - mb3};
#pragma unroll
      for (int i = 0; i < 4; ++i) {
        unsigned hi, lo;
        split_pack(xa0[i], xa1[i], hi, lo);
        *reinterpret_cast<unsigned*>(&lAhi[(c4 + i) * CST + k0]) = hi;
        *reinterpret_cast<unsigned*>(&lAlo[(c4 + i) * CST + k0]) = lo;
        split_pack(xb0[i], xb1[i], hi, lo);
        *reinterpret_cast<unsigned*>(&lBhi[(c4 + i) * CST + k0]) = hi;
        *reinterpret_cast<unsigned*>(&lBlo[(c4 + i) * CST + k0]) = lo;
      }
    }
    __syncthreads();

    // prefetch next K-chunk (global_prefetch_b8); whole input is L2-resident anyway
    if (t0 + KC < T_) {
      const int q = tid >> 1, pk = q >> 2, ln = (q & 3) << 5;
      const float* ps = ((tid & 1) ? srcB : srcA) + (size_t)(t0 + KC + pk) * N_ + ln;
      __builtin_prefetch(ps, 0, 0);
    }

    // ---- one K=32 step: bf16x3 -> 24 V_WMMA_F32_16X16X32_BF16 per wave ----
    v16bf bHi[2], bLo[2];
#pragma unroll
    for (int j = 0; j < 2; ++j) {
      const int n = waveN + j * 16 + l15;
      bHi[j] = load_frag(&lBhi[n * CST + K0B], 16);
      bLo[j] = load_frag(&lBlo[n * CST + K0B], 16);
    }
#pragma unroll
    for (int i = 0; i < 4; ++i) {
      const int m = waveM + i * 16 + l15;
      const v16bf aHi = load_frag(&lAhi[m * CST + K0A], 16);
      const v16bf aLo = load_frag(&lAlo[m * CST + K0A], 16);
#pragma unroll
      for (int j = 0; j < 2; ++j) {
        acc[i][j] = __builtin_amdgcn_wmma_f32_16x16x32_bf16(
            false, aHi, false, bHi[j], (short)0, acc[i][j], false, false);
        acc[i][j] = __builtin_amdgcn_wmma_f32_16x16x32_bf16(
            false, aHi, false, bLo[j], (short)0, acc[i][j], false, false);
        acc[i][j] = __builtin_amdgcn_wmma_f32_16x16x32_bf16(
            false, aLo, false, bHi[j], (short)0, acc[i][j], false, false);
      }
    }
    __syncthreads();
  }

  // epilogue: write raw Gram tile (+ mirrored tile when off-diagonal)
  float* gb = g + (size_t)b * N_ * N_;
  const int rhalf = (lane >> 4) << 3;              // C layout: lanes16-31 hold M+8
#pragma unroll
  for (int i = 0; i < 4; ++i)
#pragma unroll
    for (int j = 0; j < 2; ++j) {
      const int col = Jbase + waveN + j * 16 + l15;
#pragma unroll
      for (int r = 0; r < 8; ++r) {
        const int row = Ibase + waveM + i * 16 + rhalf + r;
        const float v = acc[i][j][r];
        gb[(size_t)row * N_ + col] = v;
        if (mirror) gb[(size_t)col * N_ + row] = v;
      }
    }
}

// ---------------- Kernel 3: rsd[b,n] = rsqrt(G[b,n,n]) ----------------
__global__ __launch_bounds__(256) void diag_rsqrt_kernel(const float* __restrict__ g,
                                                         float* __restrict__ rsd) {
  int idx = blockIdx.x * 256 + threadIdx.x;        // [0, 64*512)
  int b = idx >> 9, n = idx & (N_ - 1);
  rsd[idx] = rsqrtf(g[(size_t)b * N_ * N_ + (size_t)n * N_ + n]);
}

// ---------------- Kernel 4: corr = clip(G * rsd_row * rsd_col) in place ----------------
__global__ __launch_bounds__(256) void normalize_kernel(float* __restrict__ g,
                                                        const float* __restrict__ rsd) {
  size_t idx = (size_t)blockIdx.x * 256 + threadIdx.x;   // float4 index
  const int b   = (int)(idx >> 16);                      // 512*512/4 = 65536 f4 per batch
  const int rem = (int)(idx & 65535);
  const int row = rem >> 7;                              // 128 f4 per row
  const int col = (rem & 127) << 2;
  const float rr = rsd[b * N_ + row];
  const float4 rc = *reinterpret_cast<const float4*>(rsd + b * N_ + col);
  float4* p = reinterpret_cast<float4*>(g + (size_t)b * N_ * N_ + (size_t)row * N_ + col);
  float4 v = *p;
  v.x = fminf(1.f, fmaxf(-1.f, v.x * rr * rc.x));
  v.y = fminf(1.f, fmaxf(-1.f, v.y * rr * rc.y));
  v.z = fminf(1.f, fmaxf(-1.f, v.z * rr * rc.z));
  v.w = fminf(1.f, fmaxf(-1.f, v.w * rr * rc.w));
  *p = v;
}

extern "C" void kernel_launch(void* const* d_in, const int* in_sizes, int n_in,
                              void* d_out, int out_size, void* d_ws, size_t ws_size,
                              hipStream_t stream) {
  const float* x = (const float*)d_in[0];
  // d_in[1]=bn_weight, d_in[2]=bn_bias: the per-sample scalar affine BN provably cancels
  // in the column-centered correlation, so they are not needed.
  (void)in_sizes; (void)n_in; (void)out_size; (void)ws_size;

  float* out  = (float*)d_out;
  float* mean = (float*)d_ws;                  // 64*512 floats
  float* rsd  = (float*)d_ws + B_ * N_;        // 64*512 floats

  col_mean_kernel<<<dim3((B_ * N_) / 256), dim3(256), 0, stream>>>(x, mean);
  gram_kernel<<<dim3(10, B_), dim3(256), 0, stream>>>(x, mean, out);
  diag_rsqrt_kernel<<<dim3((B_ * N_) / 256), dim3(256), 0, stream>>>(out, rsd);
  normalize_kernel<<<dim3((B_ * N_ * (N_ / 4)) / 256), dim3(256), 0, stream>>>(out, rsd);
}
